// AT_kernel_71794673320462
// MI455X (gfx1250) — matrix operand order repaired and checked by reading, not compiled, not run on hardware
//
#include <hip/hip_runtime.h>
#include <hip/hip_bf16.h>

typedef __attribute__((ext_vector_type(16))) _Float16 v16h;
typedef __attribute__((ext_vector_type(8)))  _Float16 h8;
typedef __attribute__((ext_vector_type(4)))  _Float16 h4;
typedef __attribute__((ext_vector_type(8)))  float    v8f;
typedef __attribute__((ext_vector_type(4)))  unsigned int v4u;
typedef __attribute__((ext_vector_type(8)))  int      v8i;
typedef __attribute__((ext_vector_type(4)))  int      v4i;

#define BB    4
#define CC    128
#define HH    56
#define WW    56
#define PP    (HH*WW)            // 3136
#define NPIX  (BB*PP)            // 12544
#define MID   32
#define OGC   512
#define K2    9
#define COUTC 128
#define NTOT  ((float)(4*K2*PP))
#define HSTR  40                 // hstage row stride in halves (keeps 16B-aligned runs)

// ---- hand-carved LDS layout for k_attn (single 61KB block) ----
#define SM_W1   0                          // 32*256 f16   = 16384 B
#define SM_W2   16384                      // 512*32 f16   = 32768 B
#define SM_HST  49152                      // 4*16*HSTR f16 = 5120 B
#define SM_SCR  54272                      // max(4KB stats, 8KB out) = 8192 B
#define SM_SIZE 62464
#define WTOT_H  (MID*2*CC + OGC*MID)       // 24576 halves = 48KB contiguous W1h|W2h

__device__ __forceinline__ v8f wmma16(v16h a, v16h b, v8f c) {
    return __builtin_amdgcn_wmma_f32_16x16x32_f16(false, a, false, b, (short)0, c, false, false);
}
__device__ __forceinline__ v16h join16(h8 lo, h8 hi) {
    return __builtin_shufflevector(lo, hi, 0,1,2,3,4,5,6,7,8,9,10,11,12,13,14,15);
}

// Tensor Data Mover: DMA a 1-row tile of `nhalves` f16 from global -> LDS offset `lds_off`.
// D# per CDNA5 ISA ch.8: group0 = {count,lds_addr,global_addr,type=2}, group1 = dims/strides.
__device__ __forceinline__ void tdm_load_halves(const void* gptr, unsigned lds_off, unsigned nhalves) {
#if __has_builtin(__builtin_amdgcn_tensor_load_to_lds)
    unsigned long long ga = (unsigned long long)gptr;
    v4u g0;
    g0[0] = 1u;                                            // count=1 (valid descriptor)
    g0[1] = lds_off;                                       // LDS byte address
    g0[2] = (unsigned)(ga & 0xFFFFFFFFu);                  // global_addr[31:0]
    g0[3] = (unsigned)((ga >> 32) & 0x01FFFFFFu) | (2u << 30); // global_addr[56:32] | type=2
    v8i g1 = {};
    g1[0] = 1 << 16;                                       // data_size=1 (2 bytes)
    g1[1] = (int)((nhalves & 0xFFFFu) << 16);              // tensor_dim0[15:0] @ bits63:48
    g1[2] = (int)((nhalves >> 16) & 0xFFFFu) | (1 << 16);  // tensor_dim0[31:16], tensor_dim1=1
    g1[3] = (int)((nhalves & 0xFFFFu) << 16);              // tile_dim0 @ bits127:112
    g1[4] = 1;                                             // tile_dim1=1
    g1[5] = (int)nhalves;                                  // tensor_dim0_stride[31:0]
    v4i z4 = {};
#if __has_include(<hip/amd_detail/amd_gfx1250_TDM.h>)
    v8i z8 = {};
    __builtin_amdgcn_tensor_load_to_lds(g0, g1, z4, z4, z8, 0);
#else
    __builtin_amdgcn_tensor_load_to_lds(g0, g1, z4, z4, 0);
#endif
#if __has_builtin(__builtin_amdgcn_s_wait_tensorcnt)
    __builtin_amdgcn_s_wait_tensorcnt(0);
#endif
#else
    (void)gptr; (void)lds_off; (void)nhalves;
#endif
}

__global__ void k_zero(float* p, int n) {
    int i = blockIdx.x * blockDim.x + threadIdx.x;
    if (i < n) p[i] = 0.f;
}

__global__ void k_cvt_w(const float* __restrict__ s, _Float16* __restrict__ d, int n4) {
    int i = blockIdx.x * blockDim.x + threadIdx.x;
    if (i < n4) {
        float4 v = ((const float4*)s)[i];
        h4 o; o[0] = (_Float16)v.x; o[1] = (_Float16)v.y; o[2] = (_Float16)v.z; o[3] = (_Float16)v.w;
        ((h4*)d)[i] = o;
    }
}

// x [B,C,H,W] f32 -> xh2 [pixel][channel] f16 (transpose + convert)
__global__ void k_cvt_xt(const float* __restrict__ x, _Float16* __restrict__ xh2) {
    int i = blockIdx.x * blockDim.x + threadIdx.x;   // 0 .. NPIX*16-1
    if (i >= NPIX * 16) return;
    int cg = (i & 15) << 3;
    int gp = i >> 4;
    int b  = gp / PP, pp = gp - b * PP;
    const float* src = x + ((size_t)b * CC + cg) * PP + pp;
    h8 o;
    #pragma unroll
    for (int c = 0; c < 8; ++c) o[c] = (_Float16)src[c * PP];
    *(h8*)(xh2 + (size_t)gp * CC + cg) = o;
}

// ---- ke = ReLU(BN1(x*Wk^T)) (f16, pixel-major) and v = BNv(x*Wv^T) (f32, pixel-major) ----
__global__ __launch_bounds__(256) void k_embed(
    const _Float16* __restrict__ xh2,
    const _Float16* __restrict__ Wkh, const _Float16* __restrict__ Wvh,
    const float* __restrict__ bn1_g, const float* __restrict__ bn1_b,
    const float* __restrict__ bn1_m, const float* __restrict__ bn1_v,
    const float* __restrict__ bnv_g, const float* __restrict__ bnv_b,
    const float* __restrict__ bnv_m, const float* __restrict__ bnv_v,
    _Float16* __restrict__ keh2, float* __restrict__ vbuf)
{
    const int lane   = threadIdx.x & 31;
    const int wave   = threadIdx.x >> 5;
    const int nlo    = lane & 15;
    const int half16 = lane >> 4;
    const int gp0    = blockIdx.x * 16;
    const int ntile  = blockIdx.y * 8 + wave;
    const int ntu    = __builtin_amdgcn_readfirstlane(ntile);   // wave-uniform scalar
    const int j      = ntu * 16 + nlo;

    float s, t;
    if (ntu < 8) {
        float inv = rsqrtf(bn1_v[j] + 1e-5f);
        s = bn1_g[j] * inv; t = bn1_b[j] - bn1_m[j] * s;
    } else {
        int jj = j - CC;
        float inv = rsqrtf(bnv_v[jj] + 1e-5f);
        s = bnv_g[jj] * inv; t = bnv_b[jj] - bnv_m[jj] * s;
    }

    const _Float16* xrow = xh2 + (size_t)(gp0 + nlo) * CC;
    const _Float16* wrow = (ntu < 8) ? (Wkh + (size_t)j * CC) : (Wvh + (size_t)(j - CC) * CC);
    const int base8 = half16 << 3;

    v8f acc = {};
    #pragma unroll
    for (int kc = 0; kc < 4; ++kc) {
        h8 lo = *(const h8*)(xrow + kc * 32 + base8);
        h8 hi = *(const h8*)(xrow + kc * 32 + base8 + 16);
        v16h af = join16(lo, hi);
        v16h bf = join16(*(const h8*)(wrow + kc * 32 + base8), *(const h8*)(wrow + kc * 32 + base8 + 16));
        acc = wmma16(af, bf, acc);
    }
    if (ntu < 8) {
        #pragma unroll
        for (int r = 0; r < 8; ++r) {
            int gp = gp0 + r + (half16 << 3);
            keh2[(size_t)gp * CC + j] = (_Float16)fmaxf(acc[r] * s + t, 0.f);
        }
    } else {
        #pragma unroll
        for (int r = 0; r < 8; ++r) {
            int gp = gp0 + r + (half16 << 3);
            vbuf[(size_t)gp * OGC + (j - CC)] = acc[r] * s + t;
        }
    }
}

// ---- recompute a = W2*ReLU(BN2(W1*[x;unfold(ke)]+b1))+b2; 4 waves split the 9 taps ----
template <bool STATS>
__global__ __launch_bounds__(128) void k_attn(
    const _Float16* __restrict__ xh2, const _Float16* __restrict__ keh2,
    const _Float16* __restrict__ W1h,           // W2h contiguous right after W1h
    const float* __restrict__ b1,
    const float* __restrict__ bn2_g, const float* __restrict__ bn2_b,
    const float* __restrict__ bn2_m, const float* __restrict__ bn2_v,
    const float* __restrict__ b2,
    const float* __restrict__ gn_g, const float* __restrict__ gn_b,
    const float* __restrict__ vbuf,
    const float* __restrict__ meanb, const float* __restrict__ rstdb,
    float* __restrict__ gsum, float* __restrict__ gsq,
    float* __restrict__ out)
{
    __shared__ __align__(16) unsigned char smem[SM_SIZE];
    _Float16* w1s = (_Float16*)(smem + SM_W1);
    _Float16* w2s = (_Float16*)(smem + SM_W2);
    float*    scr = (float*)(smem + SM_SCR);     // lds_sums (stats) / lds_out (final)

    const int lane   = threadIdx.x & 31;
    const int wave   = threadIdx.x >> 5;
    const int nlo    = lane & 15;
    const int half16 = lane >> 4;
    const int b      = blockIdx.y;
    const int pp0    = blockIdx.x * 16;
    _Float16* hst = (_Float16*)(smem + SM_HST) + wave * 16 * HSTR;

    // stage W1h|W2h (48KB contiguous) into LDS via the Tensor Data Mover
#if __has_builtin(__builtin_amdgcn_tensor_load_to_lds)
    if (wave == 0) tdm_load_halves(W1h, SM_W1, WTOT_H);
#else
    for (int i = threadIdx.x; i < WTOT_H / 2; i += 128)
        ((unsigned*)smem)[i] = ((const unsigned*)W1h)[i];
#endif
    if (STATS) { for (int i = threadIdx.x; i < 2 * OGC; i += 128) scr[i] = 0.f; }
    else       { for (int i = threadIdx.x; i < 16 * COUTC; i += 128) scr[i] = 0.f; }
    __syncthreads();

    const int m0 = nlo, m1 = 16 + nlo;
    float s20 = bn2_g[m0] * rsqrtf(bn2_v[m0] + 1e-5f);
    float t20 = bn2_b[m0] - bn2_m[m0] * s20;
    float bb0 = b1[m0];
    float s21 = bn2_g[m1] * rsqrtf(bn2_v[m1] + 1e-5f);
    float t21 = bn2_b[m1] - bn2_m[m1] * s21;
    float bb1 = b1[m1];

    const int myp = pp0 + nlo;
    const int myy = myp / WW, myx = myp - myy * WW;
    const int base8 = half16 << 3;

    // cache x-half A-fragments (identical for all taps)
    const _Float16* xrow = xh2 + (size_t)(b * PP + myp) * CC;
    v16h axc[4];
    #pragma unroll
    for (int kc = 0; kc < 4; ++kc) {
        h8 lo = *(const h8*)(xrow + kc * 32 + base8);
        h8 hi = *(const h8*)(xrow + kc * 32 + base8 + 16);
        axc[kc] = join16(lo, hi);
    }
    const v16h vzero = {};

    for (int p = wave; p < K2; p += 4) {
        const int dy = p / 3 - 1, dx = p % 3 - 1;
        const int ny = myy + dy, nx = myx + dx;
        const bool nb_ok = ((unsigned)ny < HH) && ((unsigned)nx < WW);
        const int  gnb   = b * PP + (nb_ok ? (ny * WW + nx) : myp);
        const _Float16* krow = keh2 + (size_t)gnb * CC;

        // GEMM2: h[16x32] = QK[16x256] x W1^T (B-fragments from LDS)
        v8f h0 = {}, h1 = {};
        #pragma unroll
        for (int kc = 0; kc < 8; ++kc) {
            v16h af;
            if (kc < 4) af = axc[kc];
            else {
                h8 lo = *(const h8*)(krow + (kc - 4) * 32 + base8);
                h8 hi = *(const h8*)(krow + (kc - 4) * 32 + base8 + 16);
                v16h kf = join16(lo, hi);
                af = nb_ok ? kf : vzero;
            }
            const int kcb = kc * 32 + base8;
            v16h bf0 = join16(*(const h8*)(w1s + m0 * (2 * CC) + kcb), *(const h8*)(w1s + m0 * (2 * CC) + kcb + 16));
            v16h bf1 = join16(*(const h8*)(w1s + m1 * (2 * CC) + kcb), *(const h8*)(w1s + m1 * (2 * CC) + kcb + 16));
            h0 = wmma16(af, bf0, h0);
            h1 = wmma16(af, bf1, h1);
        }
        // bias + BN2 + ReLU; D-layout -> A-layout bounce through wave-private LDS
        #pragma unroll
        for (int r = 0; r < 8; ++r) {
            int M = r + (half16 << 3);
            hst[M * HSTR + m0] = (_Float16)fmaxf(s20 * (h0[r] + bb0) + t20, 0.f);
            hst[M * HSTR + m1] = (_Float16)fmaxf(s21 * (h1[r] + bb1) + t21, 0.f);
        }
        __builtin_amdgcn_wave_barrier();
        h8 alo = *(const h8*)(hst + nlo * HSTR + base8);
        h8 ahi = *(const h8*)(hst + nlo * HSTR + base8 + 16);
        v16h ah = join16(alo, ahi);
        __builtin_amdgcn_wave_barrier();

        int noff[8]; bool nok[8];
        #pragma unroll
        for (int r = 0; r < 8; ++r) {
            int M = r + (half16 << 3);
            int pr = pp0 + M;
            int yr = pr / WW, xr = pr - yr * WW;
            int yy = yr + dy, xx = xr + dx;
            nok[r]  = ((unsigned)yy < HH) && ((unsigned)xx < WW);
            noff[r] = yy * WW + xx;
        }

        // GEMM3: a[16x512] = h[16x32] x W2^T (B-fragments from LDS)
        #pragma unroll 4
        for (int nt3 = 0; nt3 < 32; ++nt3) {
            const int og = nt3 * 16 + nlo;
            v16h bw = join16(*(const h8*)(w2s + og * MID + base8), *(const h8*)(w2s + og * MID + base8 + 16));
            v8f aacc = {};
            aacc = wmma16(ah, bw, aacc);
            const float bias = b2[og];

            if (STATS) {
                float s8 = 0.f, q8 = 0.f;
                #pragma unroll
                for (int r = 0; r < 8; ++r) {
                    float av = aacc[r] + bias;
                    s8 += av; q8 += av * av;
                }
                s8 += __shfl_xor(s8, 16, 32);       // fold the two half-waves (same og)
                q8 += __shfl_xor(q8, 16, 32);
                if (half16 == 0) {
                    atomicAdd(&scr[og], s8);
                    atomicAdd(&scr[OGC + og], q8);
                }
            } else {
                const int cout = og >> 2;
                const float mu = meanb[b * COUTC + cout];
                const float rs = rstdb[b * COUTC + cout];
                const float gg = gn_g[og], gb = gn_b[og];
                #pragma unroll
                for (int r = 0; r < 8; ++r) {
                    int M = r + (half16 << 3);
                    float Vv = nok[r] ? vbuf[(size_t)(b * PP + noff[r]) * OGC + og] : 0.f;
                    float an = (aacc[r] + bias - mu) * rs * gg + gb;
                    float prod = an * Vv;
                    prod += __shfl_xor(prod, 1, 32);  // fold the 4 gc channels of one cout
                    prod += __shfl_xor(prod, 2, 32);
                    if ((nlo & 3) == 0)
                        atomicAdd(&scr[M * COUTC + cout], prod);
                }
            }
        }
    }
    __syncthreads();

    if (STATS) {
        for (int i = threadIdx.x; i < OGC; i += 128) {
            atomicAdd(&gsum[b * COUTC + (i >> 2)], scr[i]);
            atomicAdd(&gsq [b * COUTC + (i >> 2)], scr[OGC + i]);
        }
    } else {
        for (int i = threadIdx.x; i < 16 * COUTC; i += 128) {
            int cout = i >> 4, M = i & 15;          // 16 consecutive threads -> 64B store
            out[(size_t)(b * COUTC + cout) * PP + pp0 + M] = scr[M * COUTC + cout];
        }
    }
}

__global__ void k_finalize(const float* __restrict__ gsum, const float* __restrict__ gsq,
                           float* __restrict__ meanb, float* __restrict__ rstdb)
{
    int i = blockIdx.x * blockDim.x + threadIdx.x;
    if (i < BB * COUTC) {
        float mu  = gsum[i] / NTOT;
        float var = gsq[i] / NTOT - mu * mu;
        meanb[i] = mu;
        rstdb[i] = rsqrtf(var + 1e-5f);
    }
}

extern "C" void kernel_launch(void* const* d_in, const int* in_sizes, int n_in,
                              void* d_out, int out_size, void* d_ws, size_t ws_size,
                              hipStream_t stream)
{
    const float* x     = (const float*)d_in[0];
    const float* Wk    = (const float*)d_in[1];
    const float* bn1_g = (const float*)d_in[2];
    const float* bn1_b = (const float*)d_in[3];
    const float* bn1_m = (const float*)d_in[4];
    const float* bn1_v = (const float*)d_in[5];
    const float* W1    = (const float*)d_in[6];
    const float* b1    = (const float*)d_in[7];
    const float* bn2_g = (const float*)d_in[8];
    const float* bn2_b = (const float*)d_in[9];
    const float* bn2_m = (const float*)d_in[10];
    const float* bn2_v = (const float*)d_in[11];
    const float* W2    = (const float*)d_in[12];
    const float* b2    = (const float*)d_in[13];
    const float* gn_g  = (const float*)d_in[14];
    const float* gn_b  = (const float*)d_in[15];
    const float* Wv    = (const float*)d_in[16];
    const float* bnv_g = (const float*)d_in[17];
    const float* bnv_b = (const float*)d_in[18];
    const float* bnv_m = (const float*)d_in[19];
    const float* bnv_v = (const float*)d_in[20];
    float* out = (float*)d_out;

    _Float16* xh2  = (_Float16*)d_ws;                 // [NPIX][CC]
    _Float16* keh2 = xh2 + (size_t)NPIX * CC;         // [NPIX][CC]
    _Float16* Wkh  = keh2 + (size_t)NPIX * CC;        // [CC][CC]
    _Float16* Wvh  = Wkh + CC * CC;                   // [OGC][CC]
    _Float16* W1h  = Wvh + OGC * CC;                  // [MID][2*CC]   (W2h contiguous after)
    _Float16* W2h  = W1h + MID * 2 * CC;              // [OGC][MID]
    float*    vbuf = (float*)(W2h + OGC * MID);       // [NPIX][OGC]
    float*    gsum = vbuf + (size_t)NPIX * OGC;
    float*    gsq  = gsum + BB * COUTC;
    float*    mnb  = gsq  + BB * COUTC;
    float*    rsb  = mnb  + BB * COUTC;

    k_zero<<<dim3(4), dim3(256), 0, stream>>>(gsum, 2 * BB * COUTC);

    k_cvt_xt<<<dim3(NPIX * 16 / 256), dim3(256), 0, stream>>>(x, xh2);
    k_cvt_w<<<dim3((CC * CC / 4 + 255) / 256), dim3(256), 0, stream>>>(Wk, Wkh, CC * CC / 4);
    k_cvt_w<<<dim3((OGC * CC / 4 + 255) / 256), dim3(256), 0, stream>>>(Wv, Wvh, OGC * CC / 4);
    k_cvt_w<<<dim3((MID * 2 * CC / 4 + 255) / 256), dim3(256), 0, stream>>>(W1, W1h, MID * 2 * CC / 4);
    k_cvt_w<<<dim3((OGC * MID / 4 + 255) / 256), dim3(256), 0, stream>>>(W2, W2h, OGC * MID / 4);

    k_embed<<<dim3(NPIX / 16, 5), dim3(256), 0, stream>>>(
        xh2, Wkh, Wvh, bn1_g, bn1_b, bn1_m, bn1_v, bnv_g, bnv_b, bnv_m, bnv_v,
        keh2, vbuf);

    k_attn<true><<<dim3(PP / 16, BB), dim3(128), 0, stream>>>(
        xh2, keh2, W1h, b1, bn2_g, bn2_b, bn2_m, bn2_v, b2, gn_g, gn_b,
        vbuf, mnb, rsb, gsum, gsq, out);

    k_finalize<<<dim3(2), dim3(256), 0, stream>>>(gsum, gsq, mnb, rsb);

    k_attn<false><<<dim3(PP / 16, BB), dim3(128), 0, stream>>>(
        xh2, keh2, W1h, b1, bn2_g, bn2_b, bn2_m, bn2_v, b2, gn_g, gn_b,
        vbuf, mnb, rsb, gsum, gsq, out);

    (void)in_sizes; (void)n_in; (void)out_size; (void)ws_size;
}
